// TimeGANGenerator_3753801416937
// MI455X (gfx1250) — matrix operand order repaired and checked by reading, not compiled, not run on hardware
//
#include <hip/hip_runtime.h>
#include <hip/hip_bf16.h>

// MI455X / gfx1250, wave32. bf16 WMMA, f32 accumulate.
// WG = 32 batch rows (2 M-subtiles share every weight fragment -> 2x L2 intensity),
// 4 waves split the gate-column (N) tiles -> no duplicated weight loads in a WG.

typedef __attribute__((ext_vector_type(16))) __bf16 v16bf;
typedef __attribute__((ext_vector_type(8)))  float  v8f;

#define HD 256   // hidden
#define ID 64    // input/output feature dim
#define OD 64
#define TT 256   // seq len

// ---------- bf16 helpers ----------
__device__ __forceinline__ unsigned short f2bf(float f) {
    unsigned u = __builtin_bit_cast(unsigned, f);
    u += 0x7FFFu + ((u >> 16) & 1u);   // round-to-nearest-even
    return (unsigned short)(u >> 16);
}
__device__ __forceinline__ float bf2f(unsigned short h) {
    unsigned u = ((unsigned)h) << 16;
    return __builtin_bit_cast(float, u);
}
// fast gate activations: one v_exp each
__device__ __forceinline__ float sigmoidf(float x) {
    return 1.f / (1.f + __expf(-x));
}
__device__ __forceinline__ float fast_tanhf(float x) {
    return 1.f - 2.f / (__expf(2.f * x) + 1.f);
}

union FragU { v16bf v; int4 q[2]; };

// A fragment: 16x32 bf16 from LDS, row-major [16][ld].
// ISA 7.12.2: lane L -> row M=L%16; lanes<16: K {0..7,16..23}, lanes>=16: K {8..15,24..31}.
__device__ __forceinline__ v16bf load_a_frag(const unsigned short* src, int ld, int ktile, int lane) {
    int m  = lane & 15;
    int kb = ktile * 32 + ((lane & 16) ? 8 : 0);
    const unsigned short* p = src + m * ld + kb;
    FragU u;
    u.q[0] = *(const int4*)(p);        // K = kb .. kb+7
    u.q[1] = *(const int4*)(p + 16);   // K = kb+16 .. kb+23
    return u.v;
}

// B fragment: 32x16 bf16 of W^T; W row-major [N][K] in global (L2-hot).
// lane L -> col N=L%16; lanes<16: K {0..7,16..23}, lanes>=16: K {8..15,24..31}.
__device__ __forceinline__ v16bf load_b_frag(const unsigned short* W, int ldk, int ntile, int ktile, int lane) {
    int n  = lane & 15;
    int kb = ktile * 32 + ((lane & 16) ? 8 : 0);
    const unsigned short* p = W + (size_t)(ntile * 16 + n) * ldk + kb;
    FragU u;
    u.q[0] = *(const int4*)(p);
    u.q[1] = *(const int4*)(p + 16);
    return u.v;
}

#define WMMA_BF16(A, B, C) \
    __builtin_amdgcn_wmma_f32_16x16x32_bf16(false, (A), false, (B), (short)0, (C), false, false)

// ---------- one GRU layer: this wave's 4 column tiles, both 16-row subtiles ----------
// MUST inline so InferAddressSpaces can turn LDS pointers into ds_* and weight
// pointers into global_* (non-inlined generic params regress to flat_load).
template<int KXT>
__device__ __forceinline__ void gru_layer(
        const unsigned short* xsrc, int ldx,
        const unsigned short* hsrc, unsigned short* hdst,
        const unsigned short* Wih, int ldWih,
        const unsigned short* Whh,
        const float* __restrict__ bih, const float* __restrict__ bhh,
        int jbase, int lane)
{
    const int n     = lane & 15;
    const int mbase = (lane & 16) ? 8 : 0;
#pragma unroll 1
    for (int jj = 0; jj < 4; ++jj) {
        const int j = jbase + jj;
        v8f ar[2] = {{}, {}}, az[2] = {{}, {}}, ai[2] = {{}, {}}, ah[2] = {{}, {}};
        for (int kt = 0; kt < KXT; ++kt) {                 // input-side GEMM
            v16bf a0 = load_a_frag(xsrc,            ldx, kt, lane);
            v16bf a1 = load_a_frag(xsrc + 16 * ldx, ldx, kt, lane);
            v16bf br = load_b_frag(Wih, ldWih, j,      kt, lane);
            v16bf bz = load_b_frag(Wih, ldWih, j + 16, kt, lane);
            v16bf bi = load_b_frag(Wih, ldWih, j + 32, kt, lane);
            ar[0] = WMMA_BF16(a0, br, ar[0]);  ar[1] = WMMA_BF16(a1, br, ar[1]);
            az[0] = WMMA_BF16(a0, bz, az[0]);  az[1] = WMMA_BF16(a1, bz, az[1]);
            ai[0] = WMMA_BF16(a0, bi, ai[0]);  ai[1] = WMMA_BF16(a1, bi, ai[1]);
        }
        for (int kt = 0; kt < 8; ++kt) {                   // recurrent GEMM, K=256
            v16bf a0 = load_a_frag(hsrc,           HD, kt, lane);
            v16bf a1 = load_a_frag(hsrc + 16 * HD, HD, kt, lane);
            v16bf br = load_b_frag(Whh, HD, j,      kt, lane);
            v16bf bz = load_b_frag(Whh, HD, j + 16, kt, lane);
            v16bf bh = load_b_frag(Whh, HD, j + 32, kt, lane);
            ar[0] = WMMA_BF16(a0, br, ar[0]);  ar[1] = WMMA_BF16(a1, br, ar[1]);
            az[0] = WMMA_BF16(a0, bz, az[0]);  az[1] = WMMA_BF16(a1, bz, az[1]);
            ah[0] = WMMA_BF16(a0, bh, ah[0]);  ah[1] = WMMA_BF16(a1, bh, ah[1]);
        }
        const int col = j * 16 + n;
        float brv = bih[col]          + bhh[col];
        float bzv = bih[HD + col]     + bhh[HD + col];
        float bin = bih[2 * HD + col];
        float bhn = bhh[2 * HD + col];
#pragma unroll
        for (int s = 0; s < 2; ++s) {
#pragma unroll
            for (int r = 0; r < 8; ++r) {                  // C layout: M = s*16+mbase+r, N = col
                int m = s * 16 + mbase + r;
                float rg = sigmoidf(ar[s][r] + brv);
                float zg = sigmoidf(az[s][r] + bzv);
                float ng = fast_tanhf(ai[s][r] + bin + rg * (ah[s][r] + bhn));
                float hp = bf2f(hsrc[m * HD + col]);
                hdst[m * HD + col] = f2bf((1.f - zg) * ng + zg * hp);
            }
        }
    }
}

// ---------- WG-shared LDS layout (bf16 elements) ----------
#define LDS_H_ELEMS (3 * 2 * 32 * HD)                 // 3 layers x ping-pong x 32x256 = 49152
#define LDS_X_OFF   (LDS_H_ELEMS)                     // [32][64]
#define LDS_MLP_OFF (LDS_H_ELEMS + 32 * ID)           // [32][256]
#define LDS_ELEMS   (LDS_H_ELEMS + 32 * ID + 32 * HD) // 59392 elems = 116 KB

__global__ __launch_bounds__(128, 1)
void timegan_kernel(const float* __restrict__ noise,
                    const float* __restrict__ b_ih0, const float* __restrict__ b_hh0,
                    const float* __restrict__ b_ih12, const float* __restrict__ b_hh12,
                    const float* __restrict__ b1, const float* __restrict__ b2,
                    const unsigned short* __restrict__ wb,   // bf16 weights in d_ws
                    float* __restrict__ out)                 // [B, T, O] f32
{
    extern __shared__ unsigned short smem[];
    const int lane    = threadIdx.x & 31;
    const int wave    = threadIdx.x >> 5;     // 4 waves split the N tiles
    const int jbase   = wave * 4;
    const int rowbase = blockIdx.x * 32;      // 32 batch rows per WG

    unsigned short* xbuf = smem + LDS_X_OFF;  // [32][64]  bf16
    unsigned short* mlpb = smem + LDS_MLP_OFF;// [32][256] bf16
    auto hbuf = [&](int l, int b) { return smem + l * (2 * 32 * HD) + b * (32 * HD); };

    // bf16 weight blocks inside d_ws (element offsets)
    const unsigned short* Wih0 = wb;                  // [768][64]
    const unsigned short* Whh0 = wb + 49152;          // [768][256]
    const unsigned short* Wih1 = wb + 245760;         // [768][256]
    const unsigned short* Wih2 = wb + 442368;         // [768][256]
    const unsigned short* Whh1 = wb + 638976;         // [768][256]
    const unsigned short* Whh2 = wb + 835584;         // [768][256]
    const unsigned short* W1   = wb + 1032192;        // [256][256]
    const unsigned short* W2   = wb + 1097728;        // [64][256]

    // init: h=0 (buffer 0), x = bf16(noise tile) — cooperative over 128 threads
    for (int idx = threadIdx.x; idx < 32 * HD; idx += 128) {
        hbuf(0, 0)[idx] = 0; hbuf(1, 0)[idx] = 0; hbuf(2, 0)[idx] = 0;
    }
    for (int idx = threadIdx.x; idx < 32 * ID; idx += 128) {
        int m = idx >> 6, k = idx & 63;
        xbuf[idx] = f2bf(noise[(size_t)(rowbase + m) * ID + k]);
    }

    const int n     = lane & 15;
    const int mbase = (lane & 16) ? 8 : 0;

#pragma unroll 1
    for (int t = 0; t < TT; ++t) {
        const int cur = t & 1, nxt = cur ^ 1;
        __syncthreads();   // xbuf (prev y) and previous-step states visible
        gru_layer<2>(xbuf, ID, hbuf(0, cur), hbuf(0, nxt), Wih0, ID, Whh0,
                     b_ih0, b_hh0, jbase, lane);
        __syncthreads();   // full h0' visible
        gru_layer<8>(hbuf(0, nxt), HD, hbuf(1, cur), hbuf(1, nxt), Wih1, HD, Whh1,
                     b_ih12, b_hh12, jbase, lane);
        __syncthreads();   // full h1' visible
        gru_layer<8>(hbuf(1, nxt), HD, hbuf(2, cur), hbuf(2, nxt), Wih2, HD, Whh2,
                     b_ih12 + 3 * HD, b_hh12 + 3 * HD, jbase, lane);
        __syncthreads();   // full h2' visible

        // MLP hidden: relu(h2' @ W1^T + b1) -> mlpb (4 column tiles per wave)
#pragma unroll 1
        for (int jj = 0; jj < 4; ++jj) {
            const int j = jbase + jj;
            v8f acc[2] = {{}, {}};
            for (int kt = 0; kt < 8; ++kt) {
                v16bf a0 = load_a_frag(hbuf(2, nxt),           HD, kt, lane);
                v16bf a1 = load_a_frag(hbuf(2, nxt) + 16 * HD, HD, kt, lane);
                v16bf b  = load_b_frag(W1, HD, j, kt, lane);
                acc[0] = WMMA_BF16(a0, b, acc[0]);
                acc[1] = WMMA_BF16(a1, b, acc[1]);
            }
            float bb = b1[j * 16 + n];
#pragma unroll
            for (int s = 0; s < 2; ++s)
#pragma unroll
                for (int r = 0; r < 8; ++r) {
                    float v = acc[s][r] + bb;
                    mlpb[(s * 16 + mbase + r) * HD + j * 16 + n] = f2bf(v > 0.f ? v : 0.f);
                }
        }
        __syncthreads();   // full MLP hidden visible

        // y = tanh(mlpb @ W2^T + b2): one 16-col tile per wave -> out + next x
        {
            const int j = wave;
            v8f acc[2] = {{}, {}};
            for (int kt = 0; kt < 8; ++kt) {
                v16bf a0 = load_a_frag(mlpb,           HD, kt, lane);
                v16bf a1 = load_a_frag(mlpb + 16 * HD, HD, kt, lane);
                v16bf b  = load_b_frag(W2, HD, j, kt, lane);
                acc[0] = WMMA_BF16(a0, b, acc[0]);
                acc[1] = WMMA_BF16(a1, b, acc[1]);
            }
            float bb = b2[j * 16 + n];
#pragma unroll
            for (int s = 0; s < 2; ++s)
#pragma unroll
                for (int r = 0; r < 8; ++r) {
                    int m = s * 16 + mbase + r;
                    float v = fast_tanhf(acc[s][r] + bb);
                    out[(size_t)(rowbase + m) * (TT * OD) + (size_t)t * OD + j * 16 + n] = v;
                    xbuf[m * ID + j * 16 + n] = f2bf(v);
                }
        }
    }
}

// ---------- weight precision pre-pass ----------
__global__ void convert_f32_bf16(const float* __restrict__ src, unsigned short* __restrict__ dst, int n) {
    int i = blockIdx.x * blockDim.x + threadIdx.x;
    if (i < n) dst[i] = f2bf(src[i]);
}

extern "C" void kernel_launch(void* const* d_in, const int* in_sizes, int n_in,
                              void* d_out, int out_size, void* d_ws, size_t ws_size,
                              hipStream_t stream)
{
    const float* noise  = (const float*)d_in[0];
    const float* W_ih0  = (const float*)d_in[1];
    const float* W_hh0  = (const float*)d_in[2];
    const float* b_ih0  = (const float*)d_in[3];
    const float* b_hh0  = (const float*)d_in[4];
    const float* W_ih12 = (const float*)d_in[5];   // [2][768][256] contiguous
    const float* W_hh12 = (const float*)d_in[6];   // [2][768][256] contiguous
    const float* b_ih12 = (const float*)d_in[7];
    const float* b_hh12 = (const float*)d_in[8];
    const float* W1     = (const float*)d_in[9];
    const float* b1     = (const float*)d_in[10];
    const float* W2     = (const float*)d_in[11];
    const float* b2     = (const float*)d_in[12];

    unsigned short* wb = (unsigned short*)d_ws;
    float* out = (float*)d_out;

    auto cv = [&](const float* s, size_t off, int n) {
        convert_f32_bf16<<<(n + 255) / 256, 256, 0, stream>>>(s, wb + off, n);
    };
    cv(W_ih0,       0,  49152);
    cv(W_hh0,   49152, 196608);
    cv(W_ih12, 245760, 393216);   // layer1 @245760, layer2 @442368
    cv(W_hh12, 638976, 393216);   // layer1 @638976, layer2 @835584
    cv(W1,    1032192,  65536);
    cv(W2,    1097728,  16384);

    const size_t lds_bytes = (size_t)LDS_ELEMS * sizeof(unsigned short); // ~116 KB / WG
    timegan_kernel<<<2048 / 32, 128, lds_bytes, stream>>>(
        noise, b_ih0, b_hh0, b_ih12, b_hh12, b1, b2, wb, out);
}
